// BatchedStructureDecoder_39204461478850
// MI455X (gfx1250) — matrix operand order repaired and checked by reading, not compiled, not run on hardware
//
#include <hip/hip_runtime.h>

// ---------------------------------------------------------------------------
// adj[g] = sigmoid(Z_g @ Z_g^T),  Z_g : 512x256 fp32, G=128  (batch is
// contiguous with exactly 512 nodes/graph, so scatter+mask == reshape).
//
// MI455X roofline: 17.2 GFLOP vs ~192 MB HBM traffic (8.3 us @ 23.3 TB/s).
// bf16x2 split GEMM (acc += hi*hi + hi*lo + lo*hi in f32) runs ~3x17.2 GFLOP
// at dense-bf16 WMMA rate (~10 us) -> near the balance point; fp32 WMMA
// (16x16x4) would be ~8x slower per instruction and far off the roofline.
//
//   * 128x128 output tile / WG, full K=256 resident in LDS (264 KB of
//     320 KB/WGP), row stride 528 B -> conflict-free ds_load_b128 fragments
//     matching the ISA bf16 A(16x32)/B(32x16) VGPR layouts.
//   * 8 wave32 waves / WG, wave tile 64x32 = 8 accumulator tiles.
//   * v16bf/u32x4 ext-vector types (16-B aligned) so LDS traffic lowers to
//     ds_load_b128 / ds_store_b128 instead of split b32 pairs.
// ---------------------------------------------------------------------------

typedef __attribute__((ext_vector_type(16))) __bf16       v16bf;
typedef __attribute__((ext_vector_type(8)))  float        v8f;
typedef __attribute__((ext_vector_type(4)))  unsigned int u32x4;  // 16-B aligned
typedef __attribute__((ext_vector_type(4)))  float        f32x4;  // 16-B aligned

constexpr int G  = 128;   // graphs
constexpr int N  = 512;   // nodes per graph
constexpr int D  = 256;   // latent dim (K of the GEMM)
constexpr int BM = 128;   // block tile rows
constexpr int BN = 128;   // block tile cols
constexpr int LDS_STRIDE = D * 2 + 16;        // 528 B/row (bank-conflict pad)
constexpr int LDS_ARR    = BM * LDS_STRIDE;   // 67,584 B per array

union FragCast { u32x4 u[2]; v16bf v; };

// A fragment (16x32 bf16, MxK): lane l -> row m=l&15, kbase=(l>>4)*8.
// VGPR 0..3 = K kbase..kbase+7 (16 contiguous bytes), VGPR 4..7 = K+16.
__device__ inline v16bf loadA(const char* arr, int row0, int k0, int half, int m) {
    const char* p = arr + (size_t)(row0 + m) * LDS_STRIDE + (k0 + half * 8) * 2;
    FragCast f;
    f.u[0] = *(const u32x4*)(p);        // ds_load_b128
    f.u[1] = *(const u32x4*)(p + 32);   // ds_load_b128
    return f.v;
}

// B fragment (32x16 bf16, KxN): lane l -> col n=l&15, kbase=(l>>4)*8.
// VGPR 0..3 = K kbase..kbase+7 (16 contiguous bytes), VGPR 4..7 = K+16.
__device__ inline v16bf loadB(const char* arr, int col0, int k0, int half, int n) {
    const char* p = arr + (size_t)(col0 + n) * LDS_STRIDE + (k0 + half * 8) * 2;
    FragCast f;
    f.u[0] = *(const u32x4*)(p);        // ds_load_b128
    f.u[1] = *(const u32x4*)(p + 32);   // ds_load_b128
    return f.v;
}

__global__ void __launch_bounds__(256, 1)
gram_sigmoid_bf16x2_kernel(const float* __restrict__ z, float* __restrict__ out) {
    extern __shared__ char smem[];
    char* Ahi = smem;
    char* Alo = smem + 1 * LDS_ARR;
    char* Bhi = smem + 2 * LDS_ARR;
    char* Blo = smem + 3 * LDS_ARR;

    constexpr int TPG = (N / BM) * (N / BN);          // 16 tiles per graph
    const int g    = blockIdx.x / TPG;
    const int tile = blockIdx.x % TPG;
    const int brow = (tile / (N / BN)) * BM;
    const int bcol = (tile % (N / BN)) * BN;

    const float* zg  = z + (size_t)g * N * D;
    const int    tid = threadIdx.x;

    // ---- Phase 1: global fp32 -> LDS bf16 hi/lo (A rows, B rows) ----------
    // 8 floats per thread per step: 2x global_load_b128 in,
    // 1x ds_store_b128 to hi + 1x ds_store_b128 to lo out.
    #pragma unroll
    for (int pass = 0; pass < 2; ++pass) {
        const float* src = zg + (size_t)(pass ? bcol : brow) * D;
        char* hi = pass ? Bhi : Ahi;
        char* lo = pass ? Blo : Alo;
        // 128 rows * 32 float8 = 4096 chunks; 16 iters * 256 threads
        for (int i = 0; i < (BM * (D / 8)) / 256; ++i) {
            int e   = tid + i * 256;        // coalesced float8 index
            int row = e >> 5;               // D/8 = 32 chunks per row
            int k8  = e & 31;
            const f32x4* rp = (const f32x4*)(src + (size_t)row * D + k8 * 8);
            f32x4 a = rp[0];
            f32x4 b = rp[1];
            union { __bf16 b[8]; u32x4 u; } h, l;
            float xs[8] = {a.x, a.y, a.z, a.w, b.x, b.y, b.z, b.w};
            #pragma unroll
            for (int j = 0; j < 8; ++j) {
                __bf16 hb = (__bf16)xs[j];                 // RNE
                h.b[j] = hb;
                l.b[j] = (__bf16)(xs[j] - (float)hb);      // residual
            }
            *(u32x4*)(hi + (size_t)row * LDS_STRIDE + k8 * 16) = h.u;
            *(u32x4*)(lo + (size_t)row * LDS_STRIDE + k8 * 16) = l.u;
        }
    }
    __syncthreads();

    // ---- Phase 2: WMMA over full K (no further barriers) -------------------
    const int wave = tid >> 5;        // 0..7
    const int lane = tid & 31;
    const int wm   = wave >> 2;       // 0..1  -> 64-row band
    const int wn   = wave & 3;        // 0..3  -> 32-col band
    const int half = lane >> 4;       // 0/1
    const int l16  = lane & 15;
    const int waveRow = wm * 64;
    const int waveCol = wn * 32;

    v8f acc[4][2];
    #pragma unroll
    for (int i = 0; i < 4; ++i)
        #pragma unroll
        for (int j = 0; j < 2; ++j)
            acc[i][j] = (v8f){0.f, 0.f, 0.f, 0.f, 0.f, 0.f, 0.f, 0.f};

    #pragma unroll 2
    for (int k0 = 0; k0 < D; k0 += 32) {
        v16bf ah[4], al[4], bh[2], bl[2];
        #pragma unroll
        for (int i = 0; i < 4; ++i) {
            ah[i] = loadA(Ahi, waveRow + i * 16, k0, half, l16);
            al[i] = loadA(Alo, waveRow + i * 16, k0, half, l16);
        }
        #pragma unroll
        for (int j = 0; j < 2; ++j) {
            bh[j] = loadB(Bhi, waveCol + j * 16, k0, half, l16);
            bl[j] = loadB(Blo, waveCol + j * 16, k0, half, l16);
        }
        #pragma unroll
        for (int i = 0; i < 4; ++i)
            #pragma unroll
            for (int j = 0; j < 2; ++j) {
                acc[i][j] = __builtin_amdgcn_wmma_f32_16x16x32_bf16(
                    false, ah[i], false, bh[j], (short)0, acc[i][j], false, false);
                acc[i][j] = __builtin_amdgcn_wmma_f32_16x16x32_bf16(
                    false, ah[i], false, bl[j], (short)0, acc[i][j], false, false);
                acc[i][j] = __builtin_amdgcn_wmma_f32_16x16x32_bf16(
                    false, al[i], false, bh[j], (short)0, acc[i][j], false, false);
            }
    }

    // ---- Phase 3: sigmoid + store (C/D layout: VGPR r -> M = half*8 + r,
    //               lane&15 -> N), coalesced 64B per half-wave row ----------
    float* og = out + (size_t)g * N * N;
    #pragma unroll
    for (int i = 0; i < 4; ++i)
        #pragma unroll
        for (int j = 0; j < 2; ++j) {
            const int rbase = brow + waveRow + i * 16 + half * 8;
            const int cbase = bcol + waveCol + j * 16 + l16;
            #pragma unroll
            for (int r = 0; r < 8; ++r) {
                float x = acc[i][j][r];
                float s = 1.0f / (1.0f + __expf(-x));
                og[(size_t)(rbase + r) * N + cbase] = s;
            }
        }
}

extern "C" void kernel_launch(void* const* d_in, const int* in_sizes, int n_in,
                              void* d_out, int out_size, void* d_ws, size_t ws_size,
                              hipStream_t stream) {
    (void)in_sizes; (void)n_in; (void)d_ws; (void)ws_size; (void)out_size;
    const float* z   = (const float*)d_in[0];   // [G*N, D] fp32
    float*       out = (float*)d_out;           // [G, N, N] fp32
    const dim3 grid(G * (N / BM) * (N / BN));   // 2048 workgroups
    const dim3 block(256);                      // 8 wave32 waves
    const size_t shmem = 4 * (size_t)LDS_ARR;   // 270,336 B (<= 320 KB/WGP)
    gram_sigmoid_bf16x2_kernel<<<grid, block, shmem, stream>>>(z, out);
}
